// Conv_LSTM_10505490006477
// MI455X (gfx1250) — matrix operand order repaired and checked by reading, not compiled, not run on hardware
//
#include <hip/hip_runtime.h>
#include <hip/hip_bf16.h>
#include <stddef.h>

#define Tn 270
#define Bn 128
#define HL 256           // LSTM hidden
#define XSP 4096         // 64*64 spatial
#define XROW 4104        // 8 + 4096

typedef __attribute__((ext_vector_type(16))) __bf16 v16bf;
typedef __attribute__((ext_vector_type(8)))  __bf16 v8bf;
typedef __attribute__((ext_vector_type(8)))  float  v8f;

__device__ __forceinline__ float sigf(float x)   { return 1.0f / (1.0f + __expf(-x)); }
__device__ __forceinline__ float tanhf_(float x) { return 2.0f * sigf(2.0f * x) - 1.0f; }

// ---------------------------------------------------------------------------
// Kernel 0: zero the grid-barrier counter (must be re-zeroed every launch)
// ---------------------------------------------------------------------------
__global__ void zero_counter_kernel(unsigned* counter) {
    if (threadIdx.x == 0) counter[0] = 0u;
}

// ---------------------------------------------------------------------------
// Kernel 1: per-(b,t) CNN -> inp16[t][b][16] = concat(x_ns, fc_out)
// One 256-thread block per image.
// ---------------------------------------------------------------------------
__global__ void __launch_bounds__(256) conv_fc_kernel(
    const float* __restrict__ x,
    const float* __restrict__ w1, const float* __restrict__ b1,
    const float* __restrict__ w2, const float* __restrict__ b2,
    const float* __restrict__ fcw, const float* __restrict__ fcb,
    float* __restrict__ inp16)
{
    __shared__ float img[64 * 65];            // padded rows
    __shared__ float p1[2 * 15 * 16];         // pooled conv1, padded
    __shared__ float p2[36];                  // pooled conv2 (c*9+y*3+x)
    __shared__ float w1s[18], b1s[2], w2s[72], b2s[4], fws[288], fbs[8];

    const int tid = threadIdx.x;
    const int bt  = blockIdx.x;               // = b*T + t
    const int b   = bt / Tn;
    const int t   = bt % Tn;
    const float* xrow = x + (size_t)bt * XROW;
    float* dst = inp16 + ((size_t)t * Bn + b) * 16;

    // stage weights + image in LDS
    if (tid < 18) w1s[tid] = w1[tid];
    if (tid < 2)  b1s[tid] = b1[tid];
    if (tid < 72) w2s[tid] = w2[tid];
    if (tid < 4)  b2s[tid] = b2[tid];
    if (tid < 8)  fbs[tid] = fcb[tid];
    for (int i = tid; i < 288; i += 256) fws[i] = fcw[i];
    for (int i = tid; i < XSP; i += 256) img[(i >> 6) * 65 + (i & 63)] = xrow[8 + i];
    if (tid < 8) dst[tid] = xrow[tid];        // x_ns passthrough
    __syncthreads();

    // conv1(3x3,1->2) + relu + maxpool4  -> p1[2][15][16]
    for (int o = tid; o < 450; o += 256) {
        const int c = o / 225, rem = o % 225, py = rem / 15, px = rem % 15;
        float mx = -1e30f;
        #pragma unroll
        for (int dy = 0; dy < 4; ++dy) {
            const int y = py * 4 + dy;
            #pragma unroll
            for (int dx = 0; dx < 4; ++dx) {
                const int xx = px * 4 + dx;
                float s = b1s[c];
                #pragma unroll
                for (int i = 0; i < 3; ++i)
                    #pragma unroll
                    for (int jj = 0; jj < 3; ++jj)
                        s += w1s[c * 9 + i * 3 + jj] * img[(y + i) * 65 + xx + jj];
                mx = fmaxf(mx, fmaxf(s, 0.0f));
            }
        }
        p1[c * 240 + py * 16 + px] = mx;
    }
    __syncthreads();

    // conv2(3x3,2->4) + relu + maxpool4 -> p2[4][3][3]
    if (tid < 36) {
        const int c = tid / 9, py = (tid % 9) / 3, px = tid % 3;
        float mx = -1e30f;
        #pragma unroll
        for (int dy = 0; dy < 4; ++dy) {
            const int y = py * 4 + dy;
            #pragma unroll
            for (int dx = 0; dx < 4; ++dx) {
                const int xx = px * 4 + dx;
                float s = b2s[c];
                #pragma unroll
                for (int ic = 0; ic < 2; ++ic)
                    #pragma unroll
                    for (int i = 0; i < 3; ++i)
                        #pragma unroll
                        for (int jj = 0; jj < 3; ++jj)
                            s += w2s[((c * 2 + ic) * 3 + i) * 3 + jj] *
                                 p1[ic * 240 + (y + i) * 16 + xx + jj];
                mx = fmaxf(mx, fmaxf(s, 0.0f));
            }
        }
        p2[tid] = mx;
    }
    __syncthreads();

    // fc 36 -> 8
    if (tid < 8) {
        float s = fbs[tid];
        #pragma unroll
        for (int k = 0; k < 36; ++k) s += fws[tid * 36 + k] * p2[k];
        dst[8 + tid] = s;
    }
}

// ---------------------------------------------------------------------------
// Kernel 2: gates_ih[t][b][n] = inp16[t][b][:] . w_ih[n][:] + b_ih[n] + b_hh[n]
// grid = T*B*4 blocks (each block: 256 contiguous n for one (t,b))
// ---------------------------------------------------------------------------
__global__ void __launch_bounds__(256) gates_ih_kernel(
    const float* __restrict__ inp16, const float* __restrict__ w_ih,
    const float* __restrict__ b_ih,  const float* __restrict__ b_hh,
    float* __restrict__ gih)
{
    __shared__ float xs[16];
    const int bt   = blockIdx.x >> 2;
    const int nblk = blockIdx.x & 3;
    const int n    = nblk * 256 + threadIdx.x;
    if (threadIdx.x < 16) xs[threadIdx.x] = inp16[(size_t)bt * 16 + threadIdx.x];
    __syncthreads();
    float s = b_ih[n] + b_hh[n];
    const float* wr = w_ih + (size_t)n * 16;
    #pragma unroll
    for (int k = 0; k < 16; ++k) s += wr[k] * xs[k];
    gih[(size_t)bt * 1024 + n] = s;
}

// ---------------------------------------------------------------------------
// Grid-wide barrier (16 persistent workgroups)
// ---------------------------------------------------------------------------
__device__ __forceinline__ void grid_sync(unsigned* counter, unsigned target) {
    __syncthreads();
    if (threadIdx.x == 0) {
        __threadfence();
        atomicAdd(counter, 1u);
        while (atomicAdd(counter, 0u) < target) { __builtin_amdgcn_s_sleep(8); }
        __threadfence();
    }
    __syncthreads();
}

// ---------------------------------------------------------------------------
// Kernel 3: persistent LSTM recurrence. 16 WGs x 256 threads (8 waves).
// WG g owns hidden units j in [16g, 16g+16) -> gate rows {j, 256+j, 512+j, 768+j}.
// w_hh fragments live in registers (loop-invariant over t). Per step, ALL
// A-fragment and gih loads are issued before the WMMA block (enforced by a
// scheduling barrier) so L2 latency is paid once per step, not 8x.
// ---------------------------------------------------------------------------
__global__ void __launch_bounds__(256, 1) lstm_kernel(
    const float* __restrict__ gih, const float* __restrict__ w_hh,
    const float* __restrict__ h0,  const float* __restrict__ c0,
    __bf16* __restrict__ hbuf,     float* __restrict__ hs,
    unsigned* __restrict__ counter)
{
    __shared__ __align__(32) __bf16 Bt[4][16][HL];   // [gate chunk][n][k] = w_hh[n0+n][k]

    const int tid  = threadIdx.x;
    const int g    = blockIdx.x;          // 0..15
    const int wave = tid >> 5;            // M-tile (batch/16)
    const int lane = tid & 31;
    const int nlo  = lane & 15;
    const int hi   = lane >> 4;           // 0/1

    // Stage this WG's w_hh rows (64 rows x 256) into LDS as bf16.
    for (int i = tid; i < 4 * 16 * HL; i += 256) {
        const int c = i >> 12;
        const int n = (i >> 8) & 15;
        const int k = i & 255;
        Bt[c][n][k] = (__bf16)w_hh[(size_t)(c * 256 + g * 16 + n) * HL + k];
    }
    // Init hbuf parity-0 with our 16 columns of h0.
    for (int i = tid; i < Bn * 16; i += 256) {
        const int b = i >> 4, jj = g * 16 + (i & 15);
        hbuf[b * HL + jj] = (__bf16)h0[b * HL + jj];
    }
    // c-state: lane holds (batch = wave*16 + r + 8*hi, hidden j = g*16 + nlo), r=0..7
    const int j = g * 16 + nlo;
    float creg[8];
    #pragma unroll
    for (int r = 0; r < 8; ++r)
        creg[r] = c0[(wave * 16 + r + 8 * hi) * HL + j];
    __syncthreads();

    // Hoist ALL loop-invariant B fragments (w_hh) into registers:
    // 8 k-steps x 4 gate chunks x v16bf = 256 VGPRs. B-frag layout used here:
    // lane holds N = lane%16, K contiguous (lo lanes K 0..15, hi lanes 16..31).
    v16bf Bf0[8], Bf1[8], Bf2[8], Bf3[8];
    #pragma unroll
    for (int kk = 0; kk < 8; ++kk) {
        const int ko = kk * 32 + hi * 16;
        Bf0[kk] = *(const v16bf*)(&Bt[0][nlo][ko]);
        Bf1[kk] = *(const v16bf*)(&Bt[1][nlo][ko]);
        Bf2[kk] = *(const v16bf*)(&Bt[2][nlo][ko]);
        Bf3[kk] = *(const v16bf*)(&Bt[3][nlo][ko]);
    }

    unsigned bar = 1;
    grid_sync(counter, 16u * bar); ++bar;

    for (int t = 0; t < Tn; ++t) {
        const __bf16* hcur  = hbuf + (size_t)(t & 1) * (Bn * HL);
        __bf16*       hnext = hbuf + (size_t)((t + 1) & 1) * (Bn * HL);

        // Issue ALL A-fragment loads for this step (8 independent buffers, 2 x
        // b128 each). A layout: lane holds M = lane%16;
        // K = k0+16*hi..+15 (contiguous, matching B).
        union Au { v16bf v; v8bf h[2]; } A[8];
        const __bf16* arow = hcur + (size_t)(wave * 16 + nlo) * HL + hi * 16;
        #pragma unroll
        for (int kk = 0; kk < 8; ++kk) {
            A[kk].h[0] = *(const v8bf*)(arow + kk * 32);
            A[kk].h[1] = *(const v8bf*)(arow + kk * 32 + 8);
        }

        // Issue gih loads too (in-order loadcnt: they return after A, so they
        // never gate the WMMAs; consumed in the epilogue).
        float gv0[8], gv1[8], gv2[8], gv3[8];
        const size_t gb = (size_t)t * Bn * 1024;
        #pragma unroll
        for (int r = 0; r < 8; ++r) {
            const size_t go = gb + (size_t)(wave * 16 + r + 8 * hi) * 1024 + j;
            gv0[r] = gih[go];
            gv1[r] = gih[go + 256];
            gv2[r] = gih[go + 512];
            gv3[r] = gih[go + 768];
        }

        // Keep every load above, every WMMA below: pay L2 latency once.
        __builtin_amdgcn_sched_barrier(0);

        v8f acc0 = {0.f,0.f,0.f,0.f,0.f,0.f,0.f,0.f};
        v8f acc1 = acc0, acc2 = acc0, acc3 = acc0;

        #pragma unroll
        for (int kk = 0; kk < 8; ++kk) {
            acc0 = __builtin_amdgcn_wmma_f32_16x16x32_bf16(false, A[kk].v, false, Bf0[kk], (short)0, acc0, false, false);
            acc1 = __builtin_amdgcn_wmma_f32_16x16x32_bf16(false, A[kk].v, false, Bf1[kk], (short)0, acc1, false, false);
            acc2 = __builtin_amdgcn_wmma_f32_16x16x32_bf16(false, A[kk].v, false, Bf2[kk], (short)0, acc2, false, false);
            acc3 = __builtin_amdgcn_wmma_f32_16x16x32_bf16(false, A[kk].v, false, Bf3[kk], (short)0, acc3, false, false);
        }

        // Elementwise LSTM cell update; D layout: M = r + 8*hi, N = nlo.
        #pragma unroll
        for (int r = 0; r < 8; ++r) {
            const int brow = wave * 16 + r + 8 * hi;
            const float gi  = acc0[r] + gv0[r];
            const float gf  = acc1[r] + gv1[r];
            const float gg  = acc2[r] + gv2[r];
            const float go_ = acc3[r] + gv3[r];
            const float cn = sigf(gf) * creg[r] + sigf(gi) * tanhf_(gg);
            const float hn = sigf(go_) * tanhf_(cn);
            creg[r] = cn;
            hs[(size_t)t * Bn * HL + (size_t)brow * HL + j] = hn;
            hnext[brow * HL + j] = (__bf16)hn;
        }
        grid_sync(counter, 16u * bar); ++bar;
    }
}

// ---------------------------------------------------------------------------
// Kernel 4: head projection  out[b*T+t] = hs[t][b][:] . head_w + head_b
// one wave per output; 8 outputs per block
// ---------------------------------------------------------------------------
__global__ void __launch_bounds__(256) head_kernel(
    const float* __restrict__ hs, const float* __restrict__ head_w,
    const float* __restrict__ head_b, float* __restrict__ out)
{
    const int wave = threadIdx.x >> 5, lane = threadIdx.x & 31;
    const int idx = blockIdx.x * 8 + wave;              // = b*T + t
    const int b = idx / Tn, t = idx % Tn;
    const float* hrow = hs + ((size_t)t * Bn + b) * HL;
    float s = 0.0f;
    #pragma unroll
    for (int i = 0; i < 8; ++i) s += hrow[lane + 32 * i] * head_w[lane + 32 * i];
    #pragma unroll
    for (int off = 16; off > 0; off >>= 1) s += __shfl_xor(s, off, 32);
    if (lane == 0) out[idx] = s + head_b[0];
}

// ---------------------------------------------------------------------------
extern "C" void kernel_launch(void* const* d_in, const int* in_sizes, int n_in,
                              void* d_out, int out_size, void* d_ws, size_t ws_size,
                              hipStream_t stream) {
    const float* x   = (const float*)d_in[0];
    const float* w1  = (const float*)d_in[1];
    const float* b1  = (const float*)d_in[2];
    const float* w2  = (const float*)d_in[3];
    const float* b2  = (const float*)d_in[4];
    const float* fcw = (const float*)d_in[5];
    const float* fcb = (const float*)d_in[6];
    const float* wih = (const float*)d_in[7];
    const float* whh = (const float*)d_in[8];
    const float* bih = (const float*)d_in[9];
    const float* bhh = (const float*)d_in[10];
    const float* hw  = (const float*)d_in[11];
    const float* hb  = (const float*)d_in[12];
    const float* h0  = (const float*)d_in[13];
    const float* c0  = (const float*)d_in[14];
    float* out = (float*)d_out;

    char* ws = (char*)d_ws;
    const size_t sz_inp = (size_t)Tn * Bn * 16 * sizeof(float);        // 2.2 MB
    const size_t sz_gih = (size_t)Tn * Bn * 1024 * sizeof(float);      // 141.6 MB
    const size_t sz_hs  = (size_t)Tn * Bn * HL * sizeof(float);        // 35.4 MB
    unsigned* counter = (unsigned*)ws;
    float*   inp16 = (float*)(ws + 256);
    float*   gih   = (float*)(ws + 256 + sz_inp);
    float*   hs    = (float*)(ws + 256 + sz_inp + sz_gih);
    __bf16*  hbuf  = (__bf16*)(ws + 256 + sz_inp + sz_gih + sz_hs);    // 2*128*256 bf16

    zero_counter_kernel<<<1, 64, 0, stream>>>(counter);
    conv_fc_kernel<<<Bn * Tn, 256, 0, stream>>>(x, w1, b1, w2, b2, fcw, fcb, inp16);
    gates_ih_kernel<<<Tn * Bn * 4, 256, 0, stream>>>(inp16, wih, bih, bhh, gih);
    lstm_kernel<<<16, 256, 0, stream>>>(gih, whh, h0, c0, hbuf, hs, counter);
    head_kernel<<<(Bn * Tn) / 8, 256, 0, stream>>>(hs, hw, hb, out);
}
